// Encoder_51969104282054
// MI455X (gfx1250) — compile-verified, matrix operand order checked by reading, not run on hardware
//
#include <hip/hip_runtime.h>
#include <math.h>

typedef __bf16 bf16;
typedef __attribute__((ext_vector_type(16))) __bf16 v16bf;
typedef __attribute__((ext_vector_type(8)))  float  v8f;
typedef int v4i_ __attribute__((vector_size(16)));   // matches builtin param type from diagnostic

#define BB    16
#define NT_   4096
#define NK_   1024
#define W_    384
#define NH_   6
#define HD_   64
#define DEPTH_ 12
#define MROWS (BB*NK_)          // 16384
#define SCALEF 0.05103103630798288f   // 384^-0.5

#define GF_BIAS  1
#define GF_GELU  2
#define GF_RESID 4

#if defined(__HIP_DEVICE_COMPILE__) && __has_builtin(__builtin_amdgcn_global_load_async_to_lds_b128)
#define HAVE_ASYNC_LDS 1
#else
#define HAVE_ASYNC_LDS 0
#endif

static __device__ inline v8f v8f_zero() {
  v8f z;
#pragma unroll
  for (int i = 0; i < 8; i++) z[i] = 0.0f;
  return z;
}

// 16-byte global -> LDS copy: async path on CDNA5 (ASYNCcnt), sync fallback otherwise.
static __device__ inline void cp16(bf16* l, const bf16* g) {
#if HAVE_ASYNC_LDS
  __builtin_amdgcn_global_load_async_to_lds_b128((v4i_*)g, (v4i_*)l, 0, 0);
#else
  *(float4*)l = *(const float4*)g;
#endif
}

static __device__ inline void wait_async() {
#if HAVE_ASYNC_LDS
#if __has_builtin(__builtin_amdgcn_s_wait_asynccnt)
  __builtin_amdgcn_s_wait_asynccnt(0);
#else
  asm volatile("s_wait_asynccnt 0x0" ::: "memory");
#endif
#endif
}

// ---------------------------------------------------------------- cast f32->bf16, transposed
// src: (DEPTH, K, N) f32 ; dst: (DEPTH, N, K) bf16  (B-fragment friendly layout)
__global__ void k_cast_bf16_t(const float* __restrict__ s, bf16* __restrict__ d, int K, int N) {
  long l = blockIdx.y;
  long idx = (long)blockIdx.x * 256 + threadIdx.x;
  long tot = (long)K * N;
  if (idx >= tot) return;
  int k = (int)(idx / N), n = (int)(idx % N);
  d[l * tot + (long)n * K + k] = (bf16)s[l * tot + (long)k * N + n];
}

// ---------------------------------------------------------------- output fill + mask tail
__global__ void k_fill_out(const float* __restrict__ mt, const unsigned char* __restrict__ mask,
                           float* __restrict__ out, long ntot, int write_mask) {
  long i = (long)blockIdx.x * 256 + threadIdx.x;
  if (i < ntot) out[i] = mt[i % W_];
  if (write_mask && i < NT_) out[ntot + i] = mask[i] ? 1.0f : 0.0f;
}

// ---------------------------------------------------------------- patch embed
__global__ __launch_bounds__(192) void k_patch_embed(
    const float* __restrict__ img, const float* __restrict__ pls, const float* __restrict__ plb,
    const float* __restrict__ pw, const float* __restrict__ pb,
    const float* __restrict__ pl2s, const float* __restrict__ pl2b,
    const int* __restrict__ keep, float* __restrict__ h) {
  __shared__ float xln[192];
  __shared__ float rs[256], rq[256];
  int blk = blockIdx.x;
  int b = blk >> 10, k = blk & 1023;
  int t = keep[k];
  int gy = t >> 6, gx = t & 63;
  int f = threadIdx.x;                 // 0..191
  int c = f % 3, p = f / 3;
  int py = p >> 3, px = p & 7;
  float val = img[(((long)b * 3 + c) * 512 + gy * 8 + py) * 512 + gx * 8 + px];

  rs[f] = val; rq[f] = val * val;
  if (f < 64) { rs[f + 192] = 0.f; rq[f + 192] = 0.f; }
  __syncthreads();
  for (int o = 128; o > 0; o >>= 1) {
    if (f < o) { rs[f] += rs[f + o]; rq[f] += rq[f + o]; }
    __syncthreads();
  }
  float mean = rs[0] * (1.f / 192.f);
  float var = rq[0] * (1.f / 192.f) - mean * mean;
  float rstd = rsqrtf(var + 1e-5f);
  xln[f] = (val - mean) * rstd * pls[f] + plb[f];
  __syncthreads();

  float a0 = pb[f], a1 = pb[f + 192];
  for (int j = 0; j < 192; j++) {
    float xv = xln[j];
    a0 += xv * pw[(long)j * W_ + f];
    a1 += xv * pw[(long)j * W_ + f + 192];
  }
  __syncthreads();
  rs[f] = a0 + a1; rq[f] = a0 * a0 + a1 * a1;
  if (f < 64) { rs[f + 192] = 0.f; rq[f + 192] = 0.f; }
  __syncthreads();
  for (int o = 128; o > 0; o >>= 1) {
    if (f < o) { rs[f] += rs[f + o]; rq[f] += rq[f + o]; }
    __syncthreads();
  }
  float m2 = rs[0] * (1.f / 384.f);
  float v2 = rq[0] * (1.f / 384.f) - m2 * m2;
  float r2 = rsqrtf(v2 + 1e-5f);
  float* hrow = h + ((long)(b << 10) + k) * W_;
#pragma unroll
  for (int u = 0; u < 2; u++) {
    int o = f + u * 192;
    float yv = (u == 0 ? a0 : a1);
    float v = (yv - m2) * r2 * pl2s[o] + pl2b[o];
    int i = o >> 1;
    float freq = __expf(-(float)(2 * i) * (9.210340371976184f / 384.f));
    float ang = (float)t * freq;
    v += (o & 1) ? __cosf(ang) : __sinf(ang);
    hrow[o] = v;
  }
}

// ---------------------------------------------------------------- row LayerNorm -> bf16
__global__ __launch_bounds__(128) void k_layernorm(const float* __restrict__ X,
                                                   const float* __restrict__ s,
                                                   const float* __restrict__ bta,
                                                   bf16* __restrict__ Y) {
  int row = blockIdx.x, tid = threadIdx.x;
  const float* x = X + (long)row * W_;
  float v0 = x[tid], v1 = x[tid + 128], v2 = x[tid + 256];
  __shared__ float rs[128], rq[128];
  rs[tid] = v0 + v1 + v2;
  rq[tid] = v0 * v0 + v1 * v1 + v2 * v2;
  __syncthreads();
  for (int o = 64; o > 0; o >>= 1) {
    if (tid < o) { rs[tid] += rs[tid + o]; rq[tid] += rq[tid + o]; }
    __syncthreads();
  }
  float m = rs[0] * (1.f / 384.f);
  float var = rq[0] * (1.f / 384.f) - m * m;
  float r = rsqrtf(var + 1e-5f);
  bf16* y = Y + (long)row * W_;
  y[tid]       = (bf16)((v0 - m) * r * s[tid] + bta[tid]);
  y[tid + 128] = (bf16)((v1 - m) * r * s[tid + 128] + bta[tid + 128]);
  y[tid + 256] = (bf16)((v2 - m) * r * s[tid + 256] + bta[tid + 256]);
}

// ---------------------------------------------------------------- final LN + scatter to d_out
__global__ __launch_bounds__(128) void k_ln_scatter(const float* __restrict__ X,
                                                    const float* __restrict__ s,
                                                    const float* __restrict__ bta,
                                                    const int* __restrict__ keep,
                                                    float* __restrict__ out) {
  int row = blockIdx.x, tid = threadIdx.x;
  const float* x = X + (long)row * W_;
  float v0 = x[tid], v1 = x[tid + 128], v2 = x[tid + 256];
  __shared__ float rs[128], rq[128];
  rs[tid] = v0 + v1 + v2;
  rq[tid] = v0 * v0 + v1 * v1 + v2 * v2;
  __syncthreads();
  for (int o = 64; o > 0; o >>= 1) {
    if (tid < o) { rs[tid] += rs[tid + o]; rq[tid] += rq[tid + o]; }
    __syncthreads();
  }
  float m = rs[0] * (1.f / 384.f);
  float var = rq[0] * (1.f / 384.f) - m * m;
  float r = rsqrtf(var + 1e-5f);
  int b = row >> 10, k = row & 1023;
  int t = keep[k];
  float* y = out + ((long)b * NT_ + t) * W_;
  y[tid]       = (v0 - m) * r * s[tid] + bta[tid];
  y[tid + 128] = (v1 - m) * r * s[tid + 128] + bta[tid + 128];
  y[tid + 256] = (v2 - m) * r * s[tid + 256] + bta[tid + 256];
}

// ---------------------------------------------------------------- WMMA bf16 GEMM, 128x128 tile
// C = A(MxK) @ Bt(NxK, pre-transposed) [+bias] [gelu] [+= residual Cf] ; optional bf16 mirror Cb
// Double-buffered LDS staging via async global->LDS (ASYNCcnt) when available.
__global__ __launch_bounds__(256) void k_gemm_bf16(
    const bf16* __restrict__ A, const bf16* __restrict__ Bt, const float* __restrict__ bias,
    float* __restrict__ Cf, bf16* __restrict__ Cb, int M, int N, int K, int flags) {
  __shared__ __align__(16) bf16 As[2][128][40];
  __shared__ __align__(16) bf16 Bs[2][128][40];
  int n0 = blockIdx.x * 128, m0 = blockIdx.y * 128;
  int tid = threadIdx.x;
  int lane = tid & 31, wv = tid >> 5;
  int wm = wv >> 1, wn = wv & 1;            // 4x2 wave grid -> 32x64 per wave
  int lr = lane & 15, kb = (lane >> 4) << 3;
  int sr = tid >> 1, sh = (tid & 1) << 4;   // staging: 128 rows x 2 half-rows of 16 halfs

  const bf16* Ag = A  + (long)(m0 + sr) * K + sh;
  const bf16* Bg = Bt + (long)(n0 + sr) * K + sh;

  v8f acc[2][4];
#pragma unroll
  for (int i = 0; i < 2; i++)
#pragma unroll
    for (int j = 0; j < 4; j++) acc[i][j] = v8f_zero();

  // prologue: stage step 0 into buffer 0
  cp16(&As[0][sr][sh], Ag);     cp16(&As[0][sr][sh + 8], Ag + 8);
  cp16(&Bs[0][sr][sh], Bg);     cp16(&Bs[0][sr][sh + 8], Bg + 8);
  wait_async();
  __syncthreads();

  int nsteps = K >> 5;
  for (int st = 0; st < nsteps; st++) {
    int cur = st & 1, nxt = cur ^ 1;
    if (st + 1 < nsteps) {  // stage next tile while computing current
      const bf16* An = Ag + (st + 1) * 32;
      const bf16* Bn = Bg + (st + 1) * 32;
      cp16(&As[nxt][sr][sh], An); cp16(&As[nxt][sr][sh + 8], An + 8);
      cp16(&Bs[nxt][sr][sh], Bn); cp16(&Bs[nxt][sr][sh + 8], Bn + 8);
    }
    if (st + 2 < nsteps) {
      __builtin_prefetch(Ag + (st + 2) * 32, 0, 1);
      __builtin_prefetch(Bg + (st + 2) * 32, 0, 1);
    }
    v16bf af[2], bfr[4];
#pragma unroll
    for (int i = 0; i < 2; i++) {
      int r = wm * 32 + i * 16 + lr;
      ((float4*)&af[i])[0] = *(const float4*)&As[cur][r][kb];
      ((float4*)&af[i])[1] = *(const float4*)&As[cur][r][kb + 16];
    }
#pragma unroll
    for (int j = 0; j < 4; j++) {
      int n = wn * 64 + j * 16 + lr;
      ((float4*)&bfr[j])[0] = *(const float4*)&Bs[cur][n][kb];
      ((float4*)&bfr[j])[1] = *(const float4*)&Bs[cur][n][kb + 16];
    }
#pragma unroll
    for (int i = 0; i < 2; i++)
#pragma unroll
      for (int j = 0; j < 4; j++)
        acc[i][j] = __builtin_amdgcn_wmma_f32_16x16x32_bf16(
            false, af[i], false, bfr[j], (short)0, acc[i][j], false, false);
    wait_async();
    __syncthreads();
  }

#pragma unroll
  for (int i = 0; i < 2; i++)
#pragma unroll
    for (int j = 0; j < 4; j++)
#pragma unroll
      for (int r = 0; r < 8; r++) {
        int mrow = m0 + wm * 32 + i * 16 + r + ((lane >> 4) << 3);
        int ncol = n0 + wn * 64 + j * 16 + (lane & 15);
        float v = acc[i][j][r];
        if (flags & GF_BIAS) v += bias[ncol];
        if (flags & GF_GELU) v = 0.5f * v * (1.0f + erff(v * 0.70710678118f));
        long idx = (long)mrow * N + ncol;
        if (flags & GF_RESID) { v += Cf[idx]; Cf[idx] = v; }
        else if (Cf) Cf[idx] = v;
        if (Cb) Cb[idx] = (bf16)v;
      }
}

// ---------------------------------------------------------------- flash attention (value = K, per reference)
__global__ __launch_bounds__(128) void k_attention(const bf16* __restrict__ qkv,
                                                   bf16* __restrict__ o_out) {
  __shared__ __align__(16) bf16 Qs[64][72];
  __shared__ __align__(16) bf16 Ks[64][72];   // [key][hd]  (B-frags for S)
  __shared__ __align__(16) bf16 KT[64][72];   // [hd][key]  (B-frags for O)
  __shared__ __align__(16) bf16 Ps[4][16][72];
  int qblk = blockIdx.x;          // 0..15
  int bh = blockIdx.y;            // 0..95
  int b = bh / NH_, hh = bh % NH_;
  int tid = threadIdx.x, lane = tid & 31, wv = tid >> 5;
  int lr = lane & 15, kb = (lane >> 4) << 3;
  long rowbase = (long)b * NK_ + qblk * 64;
  long bbase = (long)b * NK_;
  int qcol = hh * HD_, kcol = W_ + hh * HD_;

  for (int idx = tid; idx < 512; idx += 128) {
    int r = idx >> 3, c = (idx & 7) << 3;
    *(float4*)&Qs[r][c] = *(const float4*)(qkv + (rowbase + r) * (3 * W_) + qcol + c);
  }

  v8f O[4];
#pragma unroll
  for (int j = 0; j < 4; j++) O[j] = v8f_zero();
  float mrow[8], lrow[8];
#pragma unroll
  for (int r = 0; r < 8; r++) { mrow[r] = -1e30f; lrow[r] = 0.f; }

  for (int j0 = 0; j0 < NK_; j0 += 64) {
    __syncthreads();
    for (int idx = tid; idx < 512; idx += 128) {
      int r = idx >> 3, c = (idx & 7) << 3;
      float4 v = *(const float4*)(qkv + (bbase + j0 + r) * (3 * W_) + kcol + c);
      *(float4*)&Ks[r][c] = v;
      bf16 tmp[8];
      *(float4*)&tmp[0] = v;
#pragma unroll
      for (int e = 0; e < 8; e++) KT[c + e][r] = tmp[e];
    }
    __syncthreads();

    v8f s[4];
#pragma unroll
    for (int j = 0; j < 4; j++) s[j] = v8f_zero();
#pragma unroll
    for (int ks = 0; ks < 2; ks++) {
      int off = ks * 32 + kb;
      v16bf a;
      ((float4*)&a)[0] = *(const float4*)&Qs[wv * 16 + lr][off];
      ((float4*)&a)[1] = *(const float4*)&Qs[wv * 16 + lr][off + 16];
#pragma unroll
      for (int j = 0; j < 4; j++) {
        v16bf bb;
        ((float4*)&bb)[0] = *(const float4*)&Ks[j * 16 + lr][off];
        ((float4*)&bb)[1] = *(const float4*)&Ks[j * 16 + lr][off + 16];
        s[j] = __builtin_amdgcn_wmma_f32_16x16x32_bf16(false, a, false, bb, (short)0, s[j],
                                                       false, false);
      }
    }

#pragma unroll
    for (int r = 0; r < 8; r++) {
      float v0 = s[0][r] * SCALEF, v1 = s[1][r] * SCALEF;
      float v2 = s[2][r] * SCALEF, v3 = s[3][r] * SCALEF;
      float mx = fmaxf(fmaxf(v0, v1), fmaxf(v2, v3));
      for (int off = 1; off < 16; off <<= 1) mx = fmaxf(mx, __shfl_xor(mx, off, 32));
      float mnew = fmaxf(mrow[r], mx);
      float corr = __expf(mrow[r] - mnew);
      float p0 = __expf(v0 - mnew), p1 = __expf(v1 - mnew);
      float p2 = __expf(v2 - mnew), p3 = __expf(v3 - mnew);
      float ss = p0 + p1 + p2 + p3;
      for (int off = 1; off < 16; off <<= 1) ss += __shfl_xor(ss, off, 32);
      lrow[r] = lrow[r] * corr + ss;
      mrow[r] = mnew;
#pragma unroll
      for (int j = 0; j < 4; j++) O[j][r] *= corr;
      int pr = r + ((lane >> 4) << 3);
      Ps[wv][pr][0 + lr]  = (bf16)p0;
      Ps[wv][pr][16 + lr] = (bf16)p1;
      Ps[wv][pr][32 + lr] = (bf16)p2;
      Ps[wv][pr][48 + lr] = (bf16)p3;
    }
    __syncthreads();

#pragma unroll
    for (int ks = 0; ks < 2; ks++) {
      int off = ks * 32 + kb;
      v16bf a;
      ((float4*)&a)[0] = *(const float4*)&Ps[wv][lr][off];
      ((float4*)&a)[1] = *(const float4*)&Ps[wv][lr][off + 16];
#pragma unroll
      for (int j = 0; j < 4; j++) {
        v16bf bb;
        ((float4*)&bb)[0] = *(const float4*)&KT[j * 16 + lr][off];
        ((float4*)&bb)[1] = *(const float4*)&KT[j * 16 + lr][off + 16];
        O[j] = __builtin_amdgcn_wmma_f32_16x16x32_bf16(false, a, false, bb, (short)0, O[j],
                                                       false, false);
      }
    }
  }

#pragma unroll
  for (int j = 0; j < 4; j++)
#pragma unroll
    for (int r = 0; r < 8; r++) {
      long row = rowbase + wv * 16 + r + ((lane >> 4) << 3);
      int col = hh * HD_ + j * 16 + lr;
      o_out[row * W_ + col] = (bf16)(O[j][r] / lrow[r]);
    }
}

// ---------------------------------------------------------------- host
extern "C" void kernel_launch(void* const* d_in, const int* in_sizes, int n_in,
                              void* d_out, int out_size, void* d_ws, size_t ws_size,
                              hipStream_t stream) {
  const float* image   = (const float*)d_in[0];
  const float* pe_ln_s = (const float*)d_in[1];
  const float* pe_ln_b = (const float*)d_in[2];
  const float* pe_w    = (const float*)d_in[3];
  const float* pe_b    = (const float*)d_in[4];
  const float* pe_ln2s = (const float*)d_in[5];
  const float* pe_ln2b = (const float*)d_in[6];
  const float* qkv_w   = (const float*)d_in[7];
  const float* out_w   = (const float*)d_in[8];
  const float* ln1_s   = (const float*)d_in[9];
  const float* ln1_b   = (const float*)d_in[10];
  const float* w1      = (const float*)d_in[11];
  const float* b1      = (const float*)d_in[12];
  const float* w2      = (const float*)d_in[13];
  const float* b2      = (const float*)d_in[14];
  const float* ln2_s   = (const float*)d_in[15];
  const float* ln2_b   = (const float*)d_in[16];
  const float* lnf_s   = (const float*)d_in[17];
  const float* lnf_b   = (const float*)d_in[18];
  const float* mask_tk = (const float*)d_in[19];
  const unsigned char* mask = (const unsigned char*)d_in[20];
  const int* keep_idx  = (const int*)d_in[21];

  char* ws = (char*)d_ws;
  size_t off = 0;
  auto alloc = [&](size_t bytes) -> void* {
    void* p = ws + off;
    off = (off + bytes + 255) & ~(size_t)255;
    return p;
  };
  float* h    = (float*)alloc((size_t)MROWS * W_ * 4);
  bf16* xb    = (bf16*) alloc((size_t)MROWS * W_ * 2);
  bf16* qkvb  = (bf16*) alloc((size_t)MROWS * 3 * W_ * 2);
  bf16* ao    = (bf16*) alloc((size_t)MROWS * W_ * 2);
  bf16* m1    = (bf16*) alloc((size_t)MROWS * 4 * W_ * 2);
  bf16* wq_bf = (bf16*) alloc((size_t)DEPTH_ * W_ * 3 * W_ * 2);   // (N=1152, K=384)
  bf16* wo_bf = (bf16*) alloc((size_t)DEPTH_ * W_ * W_ * 2);       // (N=384,  K=384)
  bf16* w1_bf = (bf16*) alloc((size_t)DEPTH_ * W_ * 4 * W_ * 2);   // (N=1536, K=384)
  bf16* w2_bf = (bf16*) alloc((size_t)DEPTH_ * 4 * W_ * W_ * 2);   // (N=384,  K=1536)

  // transpose-cast weights once: (l, K, N) f32 -> (l, N, K) bf16
  {
    long tq = (long)W_ * 3 * W_;
    long to = (long)W_ * W_;
    long t1 = (long)W_ * 4 * W_;
    k_cast_bf16_t<<<dim3((unsigned)((tq + 255) / 256), DEPTH_), 256, 0, stream>>>(qkv_w, wq_bf, W_, 3 * W_);
    k_cast_bf16_t<<<dim3((unsigned)((to + 255) / 256), DEPTH_), 256, 0, stream>>>(out_w, wo_bf, W_, W_);
    k_cast_bf16_t<<<dim3((unsigned)((t1 + 255) / 256), DEPTH_), 256, 0, stream>>>(w1, w1_bf, W_, 4 * W_);
    k_cast_bf16_t<<<dim3((unsigned)((t1 + 255) / 256), DEPTH_), 256, 0, stream>>>(w2, w2_bf, 4 * W_, W_);
  }

  long ntot = (long)BB * NT_ * W_;
  int write_mask = (out_size >= (int)(ntot + NT_)) ? 1 : 0;
  k_fill_out<<<(int)((ntot + 255) / 256), 256, 0, stream>>>(mask_tk, mask, (float*)d_out,
                                                            ntot, write_mask);

  k_patch_embed<<<MROWS, 192, 0, stream>>>(image, pe_ln_s, pe_ln_b, pe_w, pe_b,
                                           pe_ln2s, pe_ln2b, keep_idx, h);

  for (int l = 0; l < DEPTH_; l++) {
    k_layernorm<<<MROWS, 128, 0, stream>>>(h, ln1_s + (long)l * W_, ln1_b + (long)l * W_, xb);
    k_gemm_bf16<<<dim3(9, 128), 256, 0, stream>>>(
        xb, wq_bf + (long)l * W_ * 3 * W_, nullptr, nullptr, qkvb,
        MROWS, 3 * W_, W_, 0);
    k_attention<<<dim3(16, BB * NH_), 128, 0, stream>>>(qkvb, ao);
    k_gemm_bf16<<<dim3(3, 128), 256, 0, stream>>>(
        ao, wo_bf + (long)l * W_ * W_, nullptr, h, nullptr,
        MROWS, W_, W_, GF_RESID);
    k_layernorm<<<MROWS, 128, 0, stream>>>(h, ln2_s + (long)l * W_, ln2_b + (long)l * W_, xb);
    k_gemm_bf16<<<dim3(12, 128), 256, 0, stream>>>(
        xb, w1_bf + (long)l * W_ * 4 * W_, b1 + (long)l * 4 * W_, nullptr, m1,
        MROWS, 4 * W_, W_, GF_BIAS | GF_GELU);
    k_gemm_bf16<<<dim3(3, 128), 256, 0, stream>>>(
        m1, w2_bf + (long)l * 4 * W_ * W_, b2 + (long)l * W_, h, nullptr,
        MROWS, W_, 4 * W_, GF_BIAS | GF_GELU | GF_RESID);
  }

  k_ln_scatter<<<MROWS, 128, 0, stream>>>(h, lnf_s, lnf_b, keep_idx, (float*)d_out);
  (void)n_in; (void)in_sizes; (void)ws_size;
}
